// AttentiveTransformer_1812476199371
// MI455X (gfx1250) — matrix operand order repaired and checked by reading, not compiled, not run on hardware
//
#include <hip/hip_runtime.h>
#include <hip/hip_bf16.h>

// CDNA5 / gfx1250. wave32. v_wmma_f32_16x16x32_bf16 GEMM with
// GLOBAL_LOAD_ASYNC_TO_LDS_B128 double-buffered LDS staging (ASYNCcnt).

typedef __attribute__((ext_vector_type(16))) __bf16 v16bf;
typedef __attribute__((ext_vector_type(8)))  float  v8f;

#define VBS 128
#define BN_EPS 1e-5f

// GEMM tiling
#define TILE_K   32
#define ROWB     80                  // LDS row pitch bytes (40 halves, 16B aligned, bank-stride 20)
#define ABUF_B   (128 * ROWB)        // 10240 B per A stage
#define STAGE_B  (2 * ABUF_B)        // A + B per stage = 20480 B

// ---------------------------------------------------------------- utilities

__device__ __forceinline__ unsigned short f2bf_rne(float f) {
    unsigned u = __float_as_uint(f);
    unsigned r = u + 0x7FFFu + ((u >> 16) & 1u);   // round-to-nearest-even
    return (unsigned short)(r >> 16);
}

__device__ __forceinline__ unsigned pack2bf(float a, float b) {
    return (unsigned)f2bf_rne(a) | ((unsigned)f2bf_rne(b) << 16);
}

// fp32 -> bf16, 8 elements per thread, 32B in / 16B out
__global__ void f32_to_bf16_kernel(const float* __restrict__ in,
                                   __bf16* __restrict__ out, int n) {
    int i = (blockIdx.x * blockDim.x + threadIdx.x) * 8;
    if (i >= n) return;
    float4 f0 = *(const float4*)(in + i);
    float4 f1 = *(const float4*)(in + i + 4);
    uint4 o;
    o.x = pack2bf(f0.x, f0.y);
    o.y = pack2bf(f0.z, f0.w);
    o.z = pack2bf(f1.x, f1.y);
    o.w = pack2bf(f1.z, f1.w);
    *(uint4*)(out + i) = o;
}

// ---------------------------------------------------------------- WMMA GEMM
// C[b,g] = sum_k A[b,k] * W[g,k]   A:[Btot,K] bf16 row-major, W:[G,K] bf16
// Block: 256 threads = 8 waves arranged 2(M) x 4(N); wave tile 64x32 =
// 4x2 fragments; workgroup tile 128x128; K-step 32, 2-stage async pipeline.
//
// Fragment layouts (CDNA5 wave32, 16-bit):
//  A 16x32 (MxK): lane m<16 -> row m, K{0..7}u{16..23}; lane m+16 -> K{8..15}u{24..31}
//  B 32x16 (KxN): lane n<16 -> col n, K{0..7}u{16..23}; lane n+16 -> K{8..15}u{24..31}
//  C 16x16:       VGPR r: lanes 0-15 -> M=r, lanes 16-31 -> M=r+8; N=lane&15
__global__ __launch_bounds__(256)
void gemm_bf16_wmma_async_kernel(const __bf16* __restrict__ A,
                                 const __bf16* __restrict__ W,
                                 float* __restrict__ C,
                                 int K, int G) {
    __shared__ __align__(16) unsigned char smem[2 * STAGE_B];   // 40 KB
    const unsigned ldsBase = (unsigned)(uintptr_t)(void*)smem;  // low 32b = LDS offset

    const int lane   = threadIdx.x & 31;
    const int waveId = threadIdx.x >> 5;
    const int waveM  = waveId & 1;
    const int waveN  = waveId >> 1;
    const int m      = lane & 15;
    const int h      = lane >> 4;

    const int rowBase0 = blockIdx.y * 128;
    const int colBase0 = blockIdx.x * 128;
    const int rowBase  = rowBase0 + waveM * 64;
    const int colBase  = colBase0 + waveN * 32;

    // ---- async-copy role: waves 0-3 stage the A tile, waves 4-7 the B tile.
    // Chunk c = (cw*4 + q)*32 + lane, row = c>>2 (tile row), ch = c&3 (16B chunk).
    const bool copyA = (waveId < 4);
    const int  cw    = copyA ? waveId : waveId - 4;
    const __bf16* gsrc[4];
    unsigned ldsOff[4];
#pragma unroll
    for (int q = 0; q < 4; ++q) {
        int c   = cw * 128 + q * 32 + lane;
        int row = c >> 2;
        int ch  = c & 3;
        const __bf16* base = copyA ? (A + (size_t)(rowBase0 + row) * K)
                                   : (W + (size_t)(colBase0 + row) * K);
        gsrc[q]   = base + ch * 8;
        ldsOff[q] = (copyA ? 0u : (unsigned)ABUF_B) + row * ROWB + ch * 16;
    }

    auto issue_stage = [&](int stage, int k) {
#pragma unroll
        for (int q = 0; q < 4; ++q) {
            unsigned ldsAddr = ldsBase + stage * STAGE_B + ldsOff[q];
            unsigned long long ga = (unsigned long long)(uintptr_t)(gsrc[q] + k);
            asm volatile("global_load_async_to_lds_b128 %0, %1, off"
                         :: "v"(ldsAddr), "v"(ga) : "memory");
        }
    };

    // ---- fragment LDS byte offsets (within a stage)
    unsigned aOff[4], bOff[2];
#pragma unroll
    for (int i = 0; i < 4; ++i)
        aOff[i] = (waveM * 64 + i * 16 + m) * ROWB + h * 16;    // chunk1 at +32
#pragma unroll
    for (int j = 0; j < 2; ++j)
        bOff[j] = ABUF_B + (waveN * 32 + j * 16 + m) * ROWB + h * 16; // chunk1 at +32

    union V8 { v8f v; float f[8]; };
    V8 acc[4][2];
#pragma unroll
    for (int i = 0; i < 4; ++i)
#pragma unroll
        for (int j = 0; j < 2; ++j)
            acc[i][j].v = (v8f){0.f, 0.f, 0.f, 0.f, 0.f, 0.f, 0.f, 0.f};

    union Frag { v16bf v; uint4 q[2]; };

    const int NK = K / TILE_K;

    issue_stage(0, 0);
    asm volatile("s_wait_asynccnt 0x0" ::: "memory");
    __syncthreads();

    for (int kt = 0; kt < NK; ++kt) {
        const int cur = kt & 1;
        if (kt + 1 < NK) issue_stage(1 - cur, (kt + 1) * TILE_K);

        const unsigned char* st = smem + cur * STAGE_B;
        Frag a[4], b[2];
#pragma unroll
        for (int i = 0; i < 4; ++i) {
            a[i].q[0] = *(const uint4*)(st + aOff[i]);
            a[i].q[1] = *(const uint4*)(st + aOff[i] + 32);
        }
#pragma unroll
        for (int j = 0; j < 2; ++j) {
            b[j].q[0] = *(const uint4*)(st + bOff[j]);
            b[j].q[1] = *(const uint4*)(st + bOff[j] + 32);
        }

#pragma unroll
        for (int i = 0; i < 4; ++i)
#pragma unroll
            for (int j = 0; j < 2; ++j)
                acc[i][j].v = __builtin_amdgcn_wmma_f32_16x16x32_bf16(
                    false, a[i].v, false, b[j].v,
                    (short)0, acc[i][j].v, false, false);

        // own next-stage copies done, then whole-WGP sync (also fences DS reads)
        asm volatile("s_wait_asynccnt 0x0" ::: "memory");
        __syncthreads();
    }

#pragma unroll
    for (int i = 0; i < 4; ++i)
#pragma unroll
        for (int j = 0; j < 2; ++j) {
            float* cp = C + (size_t)(rowBase + i * 16 + h * 8) * G
                          + colBase + j * 16 + m;
#pragma unroll
            for (int r = 0; r < 8; ++r)
                cp[(size_t)r * G] = acc[i][j].f[r];
        }
}

// ------------------------------------------------------- Ghost BN statistics
__global__ void gbn_stats_kernel(const float* __restrict__ X,
                                 float* __restrict__ mean,
                                 float* __restrict__ rstd, int D) {
    const int col = blockIdx.x * blockDim.x + threadIdx.x;
    const int grp = blockIdx.y;
    const float* p = X + (size_t)grp * VBS * D + col;
    float s = 0.f, s2 = 0.f;
#pragma unroll 4
    for (int r = 0; r < VBS; ++r) {
        float v = p[(size_t)r * D];
        s += v;
        s2 = fmaf(v, v, s2);
    }
    const float inv = 1.0f / (float)VBS;
    float mu  = s * inv;
    float var = fmaxf(s2 * inv - mu * mu, 0.f);   // biased var (BN training)
    mean[(size_t)grp * D + col] = mu;
    rstd[(size_t)grp * D + col] = rsqrtf(var + BN_EPS);
}

// --------------------------------------------------------- block reductions

__device__ __forceinline__ float waveReduceSum(float v) {
#pragma unroll
    for (int o = 16; o > 0; o >>= 1) v += __shfl_down(v, o, 32);
    return v;
}
__device__ __forceinline__ float waveReduceMax(float v) {
#pragma unroll
    for (int o = 16; o > 0; o >>= 1) v = fmaxf(v, __shfl_down(v, o, 32));
    return v;
}

__device__ __forceinline__ float blockSum(float v, float* sm) {
    v = waveReduceSum(v);
    __syncthreads();
    if ((threadIdx.x & 31) == 0) sm[threadIdx.x >> 5] = v;
    __syncthreads();
    if (threadIdx.x == 0) {
        float s = sm[0];
        for (int i = 1; i < 8; ++i) s += sm[i];
        sm[0] = s;
    }
    __syncthreads();
    return sm[0];
}

__device__ __forceinline__ float blockMax(float v, float* sm) {
    v = waveReduceMax(v);
    __syncthreads();
    if ((threadIdx.x & 31) == 0) sm[threadIdx.x >> 5] = v;
    __syncthreads();
    if (threadIdx.x == 0) {
        float s = sm[0];
        for (int i = 1; i < 8; ++i) s = fmaxf(s, sm[i]);
        sm[0] = s;
    }
    __syncthreads();
    return sm[0];
}

// ---------------------------- fused BN-apply * priors -> sparsemax (bisection)
// One block (256 threads) per row; D = 2048 (8 elems/thread, in regs).
// tau solves sum(relu(z - tau)) = 1 on [max(z)-1, max(z)] (monotone).
__global__ __launch_bounds__(256)
void bn_prior_sparsemax_kernel(const float* __restrict__ X,
                               const float* __restrict__ priors,
                               const float* __restrict__ mean,
                               const float* __restrict__ rstd,
                               const float* __restrict__ gamma,
                               const float* __restrict__ beta,
                               float* __restrict__ out, int D) {
    __shared__ float sm[8];
    const int b = blockIdx.x;
    const int grp = b >> 7;                 // b / VBS
    const size_t rowOff  = (size_t)b * D;
    const size_t statOff = (size_t)grp * D;

    float z[8];
    float zmax = -INFINITY;
#pragma unroll
    for (int t = 0; t < 8; ++t) {
        int d = threadIdx.x + t * 256;
        float x  = X[rowOff + d];
        float xh = (x - mean[statOff + d]) * rstd[statOff + d];
        float v  = fmaf(gamma[d], xh, beta[d]) * priors[rowOff + d];
        z[t] = v;
        zmax = fmaxf(zmax, v);
    }
    zmax = blockMax(zmax, sm);

    float lo = zmax - 1.0f, hi = zmax;
    for (int it = 0; it < 32; ++it) {
        float mid = 0.5f * (lo + hi);
        float s = 0.f;
#pragma unroll
        for (int t = 0; t < 8; ++t) s += fmaxf(z[t] - mid, 0.f);
        s = blockSum(s, sm);
        if (s > 1.0f) lo = mid; else hi = mid;
    }
    float tau = 0.5f * (lo + hi);

#pragma unroll
    for (int t = 0; t < 8; ++t) {
        int d = threadIdx.x + t * 256;
        out[rowOff + d] = fmaxf(z[t] - tau, 0.f);
    }
}

// ------------------------------------------------------------------- driver

extern "C" void kernel_launch(void* const* d_in, const int* in_sizes, int n_in,
                              void* d_out, int out_size, void* d_ws, size_t ws_size,
                              hipStream_t stream) {
    const float* priors = (const float*)d_in[0];   // [B, G]
    const float* feat   = (const float*)d_in[1];   // [B, K]
    const float* fcw    = (const float*)d_in[2];   // [G, K]
    const float* gamma  = (const float*)d_in[3];   // [G]
    const float* beta   = (const float*)d_in[4];   // [G]

    const int G = in_sizes[3];                     // 2048
    const int K = in_sizes[2] / G;                 // 2048
    const int B = in_sizes[1] / K;                 // 16384
    const int nGroups = B / VBS;                   // 128

    float* out = (float*)d_out;                    // [B, G]; reused as X buffer

    // workspace: feat_bf16 | w_bf16 | mean | rstd  (~78 MB)
    char* ws = (char*)d_ws;
    __bf16* featb = (__bf16*)ws;
    size_t featBytes = (size_t)B * K * sizeof(__bf16);
    __bf16* wb = (__bf16*)(ws + featBytes);
    size_t wBytes = (size_t)G * K * sizeof(__bf16);
    float* meanp = (float*)(ws + featBytes + wBytes);
    float* rstdp = meanp + (size_t)nGroups * G;

    // 1) fp32 -> bf16
    {
        int n = B * K;
        f32_to_bf16_kernel<<<(n + 2047) / 2048, 256, 0, stream>>>(feat, featb, n);
    }
    {
        int n = G * K;
        f32_to_bf16_kernel<<<(n + 2047) / 2048, 256, 0, stream>>>(fcw, wb, n);
    }

    // 2) async-staged WMMA GEMM -> d_out used as pre-BN activation buffer X
    gemm_bf16_wmma_async_kernel<<<dim3(G / 128, B / 128), 256, 0, stream>>>(
        featb, wb, out, K, G);

    // 3) Ghost BN statistics per (virtual batch, column)
    gbn_stats_kernel<<<dim3(G / 256, nGroups), 256, 0, stream>>>(
        out, meanp, rstdp, G);

    // 4) fused BN apply * priors -> sparsemax, in place on d_out
    bn_prior_sparsemax_kernel<<<B, 256, 0, stream>>>(
        out, priors, meanp, rstdp, gamma, beta, out, G);
}
